// PatchEmbedding_55044300866092
// MI455X (gfx1250) — compile-verified, matrix operand order checked by reading, not run on hardware
//
#include <hip/hip_runtime.h>
#include <hip/hip_bf16.h>
#include <math.h>

typedef __attribute__((ext_vector_type(16))) __bf16 v16bf;
typedef __attribute__((ext_vector_type(8)))  float  v8f;

#define PB 4
#define PN 8192
#define PT 16
#define PC 3
#define PH 256
#define PW 256
#define PPATCH 9
#define PDIM 243      /* 9*9*3 */
#define KP 256        /* padded K for GEMM1 */
#define EMB 768

__device__ __forceinline__ unsigned short f2bf(float f) {
    unsigned int u = __float_as_uint(f);
    unsigned int r = (u + 0x7FFFu + ((u >> 16) & 1u)) >> 16;  // round-to-nearest-even
    return (unsigned short)r;
}

// Branch-free erf (Abramowitz-Stegun 7.1.26, |err| <= 1.5e-7), trans-op based.
__device__ __forceinline__ float erf_fast(float x) {
    float ax = __builtin_fabsf(x);
    float t  = __builtin_amdgcn_rcpf(__builtin_fmaf(0.3275911f, ax, 1.0f));
    float p  = 1.061405429f;
    p = __builtin_fmaf(p, t, -1.453152027f);
    p = __builtin_fmaf(p, t,  1.421413741f);
    p = __builtin_fmaf(p, t, -0.284496736f);
    p = __builtin_fmaf(p, t,  0.254829592f);
    float e  = __expf(-ax * ax);
    float r  = __builtin_fmaf(-p * t, e, 1.0f);
    return (x < 0.0f) ? -r : r;
}

// Low 32 bits of a generic pointer to __shared__ = LDS byte offset.
__device__ __forceinline__ unsigned lds_addr32(const void* p) {
    return (unsigned)(unsigned long long)(p);
}

// ---------------------------------------------------------------------------
// Gather 9x9x3 patches -> bf16 A matrix [32768][256] (K padded with zeros).
// ---------------------------------------------------------------------------
__global__ __launch_bounds__(256)
void pack_x_kernel(const float* __restrict__ frames,
                   const float* __restrict__ coords,
                   const int*   __restrict__ tsrc,
                   unsigned int* __restrict__ xout) {
    int tid = blockIdx.x * blockDim.x + threadIdx.x;   // rows * 128 dwords
    int row = tid >> 7;
    int d   = tid & 127;
    int b = row >> 13;            // / 8192
    int n = row & (PN - 1);
    float cu = coords[(size_t)(b * PN + n) * 2 + 0];
    float cv = coords[(size_t)(b * PN + n) * 2 + 1];
    int u = (int)(cu * (float)(PW - 1));
    int v = (int)(cv * (float)(PH - 1));
    int t = tsrc[b * PN + n];
    unsigned int packed = 0;
#pragma unroll
    for (int half = 0; half < 2; ++half) {
        int k = 2 * d + half;
        float val = 0.0f;
        if (k < PDIM) {
            int c  = k % 3;
            int q  = k / 3;
            int dx = q % PPATCH;
            int dy = q / PPATCH;
            int y = v - PPATCH / 2 + dy;
            if (y < 0) y = 0; if (y > PH - 1) y = PH - 1;
            int x = u - PPATCH / 2 + dx;
            if (x < 0) x = 0; if (x > PW - 1) x = PW - 1;
            val = frames[(((size_t)b * PT + t) * PC + c) * (size_t)(PH * PW)
                         + (size_t)y * PW + x];
        }
        packed |= ((unsigned int)f2bf(val)) << (16 * half);
    }
    xout[(size_t)row * (KP / 2) + d] = packed;
}

// ---------------------------------------------------------------------------
// f32 row-major [rows x cols] -> bf16 TRANSPOSED [cols x rowsPad], zero-padded.
// ---------------------------------------------------------------------------
__global__ __launch_bounds__(256)
void cvt_pad_transpose_kernel(const float* __restrict__ src,
                              unsigned short* __restrict__ dst,
                              int rows_src, int rowsPad, int cols) {
    int tid = blockIdx.x * blockDim.x + threadIdx.x;   // over cols * rowsPad
    int n = tid / rowsPad;
    int k = tid - n * rowsPad;
    float v = (k < rows_src) ? src[(size_t)k * cols + n] : 0.0f;
    dst[tid] = f2bf(v);
}

// ---------------------------------------------------------------------------
// bf16 WMMA GEMM with double-buffered ASYNC global->LDS staging.
//   A  : bf16 row-major,    K contiguous  (K = Kc*32)
//   Bt : bf16 column-major ([n][k]), K contiguous -- same tiling as A
//   MODE 0: GELU (poly erf), bf16 output; MODE 1: bias only, f32 output
// Block: 256 threads = 8 wave32 waves; block tile 128(M) x 128(N);
// wave tile 32x64 = eight v_wmma_f32_16x16x32_bf16 per K chunk.
// ---------------------------------------------------------------------------
template <int MODE>
__global__ __launch_bounds__(256)
void gemm_wmma_kernel(const unsigned short* __restrict__ A,
                      const unsigned short* __restrict__ Bt,
                      const float* __restrict__ bias,
                      void* __restrict__ outp,
                      int Kc) {
    // [buffer][row/col][pitch 20 dwords]; rows are 80B (16B-aligned) apart
    __shared__ __align__(16) unsigned int Asm[2][128][20];
    __shared__ __align__(16) unsigned int Bsm[2][128][20];

    const int t    = threadIdx.x;
    const int lane = t & 31;
    const int w    = t >> 5;
    const int wm   = w & 3;       // 4 waves along M (32 rows each)
    const int wn   = w >> 2;      // 2 waves along N (64 cols each)
    const int hsel = lane >> 4;   // K-half select per ISA 16-bit A/B layout
    const int rowBase = blockIdx.x * 128;
    const int colBase = blockIdx.y * 128;
    const int Kd = Kc * 16;       // dwords per row of A and of Bt

    union Frag { v16bf v; uint4 q[2]; };
    v8f acc[2][4] = {};

    const unsigned int* Ad = (const unsigned int*)A;
    const unsigned int* Bd = (const unsigned int*)Bt;

    // Per-thread slice of the cooperative copy: 2 x b128 for A, 2 x b128 for B.
    const int r0  = t >> 2;             // rows r0 and r0+64
    const int c4  = t & 3;              // 16B column within the 64B row chunk
    const unsigned aBase0 = lds_addr32(&Asm[0][r0][c4 * 4]);
    const unsigned aBase1 = lds_addr32(&Asm[0][r0 + 64][c4 * 4]);
    const unsigned bBase0 = lds_addr32(&Bsm[0][r0][c4 * 4]);
    const unsigned bBase1 = lds_addr32(&Bsm[0][r0 + 64][c4 * 4]);
    const unsigned bufStride = (unsigned)(128 * 20 * 4);   // bytes per buffer

    auto issue_tile = [&](int kt, int p) {
        unsigned po = (unsigned)p * bufStride;
        const unsigned int* ga0 = Ad + (size_t)(rowBase + r0)      * Kd + kt * 16 + c4 * 4;
        const unsigned int* ga1 = Ad + (size_t)(rowBase + r0 + 64) * Kd + kt * 16 + c4 * 4;
        const unsigned int* gb0 = Bd + (size_t)(colBase + r0)      * Kd + kt * 16 + c4 * 4;
        const unsigned int* gb1 = Bd + (size_t)(colBase + r0 + 64) * Kd + kt * 16 + c4 * 4;
        asm volatile("global_load_async_to_lds_b128 %0, %1, off"
                     :: "v"(aBase0 + po), "v"((unsigned long long)ga0) : "memory");
        asm volatile("global_load_async_to_lds_b128 %0, %1, off"
                     :: "v"(aBase1 + po), "v"((unsigned long long)ga1) : "memory");
        asm volatile("global_load_async_to_lds_b128 %0, %1, off"
                     :: "v"(bBase0 + po), "v"((unsigned long long)gb0) : "memory");
        asm volatile("global_load_async_to_lds_b128 %0, %1, off"
                     :: "v"(bBase1 + po), "v"((unsigned long long)gb1) : "memory");
    };

    // Prologue: fill buffer 0 with tile 0.
    issue_tile(0, 0);
    asm volatile("s_wait_asynccnt 0x0" ::: "memory");
    __syncthreads();

    for (int kt = 0; kt < Kc; ++kt) {
        const int p = kt & 1;
        if (kt + 1 < Kc) issue_tile(kt + 1, p ^ 1);   // uniform branch, EXEC stays full

        // ---- fragment loads: two ds_load_b128 per fragment
        Frag a[2], b[4];
#pragma unroll
        for (int s = 0; s < 2; ++s) {
            const unsigned int* ar = &Asm[p][wm * 32 + s * 16 + (lane & 15)][0];
            a[s].q[0] = ((const uint4*)ar)[hsel];
            a[s].q[1] = ((const uint4*)(ar + 8))[hsel];
        }
#pragma unroll
        for (int s = 0; s < 4; ++s) {
            const unsigned int* br = &Bsm[p][wn * 64 + s * 16 + (lane & 15)][0];
            b[s].q[0] = ((const uint4*)br)[hsel];
            b[s].q[1] = ((const uint4*)(br + 8))[hsel];
        }
#pragma unroll
        for (int sm = 0; sm < 2; ++sm)
#pragma unroll
            for (int sn = 0; sn < 4; ++sn)
                acc[sm][sn] = __builtin_amdgcn_wmma_f32_16x16x32_bf16(
                    false, a[sm].v, false, b[sn].v, (short)0, acc[sm][sn], false, false);

        // Next tile's async writes must land before anyone reads it after the barrier.
        asm volatile("s_wait_asynccnt 0x0" ::: "memory");
        __syncthreads();
    }

    // ---- epilogue: D layout = VGPR r -> m = r + 8*hsel, n = lane & 15
#pragma unroll
    for (int sn = 0; sn < 4; ++sn) {
        const int gcol = colBase + wn * 64 + sn * 16 + (lane & 15);
        const float bv = bias[gcol];
#pragma unroll
        for (int sm = 0; sm < 2; ++sm) {
#pragma unroll
            for (int r = 0; r < 8; ++r) {
                int grow = rowBase + wm * 32 + sm * 16 + r + (hsel << 3);
                float v = acc[sm][sn][r] + bv;
                if (MODE == 0) {
                    float g = 0.5f * v * (1.0f + erf_fast(v * 0.70710678118654752440f));
                    ((unsigned short*)outp)[(size_t)grow * EMB + gcol] = f2bf(g);
                } else {
                    ((float*)outp)[(size_t)grow * EMB + gcol] = v;
                }
            }
        }
    }
}

// ---------------------------------------------------------------------------
extern "C" void kernel_launch(void* const* d_in, const int* in_sizes, int n_in,
                              void* d_out, int out_size, void* d_ws, size_t ws_size,
                              hipStream_t stream) {
    const float* frames = (const float*)d_in[0];
    const float* coords = (const float*)d_in[1];
    const int*   tsrc   = (const int*)d_in[2];
    const float* W1     = (const float*)d_in[3];
    const float* b1     = (const float*)d_in[4];
    const float* W2     = (const float*)d_in[5];
    const float* b2     = (const float*)d_in[6];

    char* ws = (char*)d_ws;
    const size_t xBytes  = (size_t)PB * PN * KP * 2;        // 16,777,216
    const size_t w1Bytes = (size_t)EMB * KP * 2;            //    393,216  (W1^T padded)
    const size_t w2Bytes = (size_t)EMB * EMB * 2;           //  1,179,648  (W2^T)
    unsigned int*   xpk = (unsigned int*)(ws);
    unsigned short* w1t = (unsigned short*)(ws + xBytes);
    unsigned short* w2t = (unsigned short*)(ws + xBytes + w1Bytes);
    unsigned short* hb  = (unsigned short*)(ws + xBytes + w1Bytes + w2Bytes);

    // 1) gather + pack A (bf16, K padded to 256)
    pack_x_kernel<<<(PB * PN * (KP / 2)) / 256, 256, 0, stream>>>(frames, coords, tsrc, xpk);
    // 2) weights -> transposed bf16 panels (W1: 243->256 K padding)
    cvt_pad_transpose_kernel<<<(EMB * KP) / 256, 256, 0, stream>>>(W1, w1t, PDIM, KP, EMB);
    cvt_pad_transpose_kernel<<<(EMB * EMB) / 256, 256, 0, stream>>>(W2, w2t, EMB, EMB, EMB);

    dim3 grid(PB * PN / 128, EMB / 128);   // (256, 6)
    // 3) GEMM1 + bias + GELU -> bf16 hidden
    gemm_wmma_kernel<0><<<grid, 256, 0, stream>>>((const unsigned short*)xpk, w1t, b1, hb,
                                                  KP / 32);
    // 4) GEMM2 + bias -> f32 output
    gemm_wmma_kernel<1><<<grid, 256, 0, stream>>>(hb, w2t, b2, d_out, EMB / 32);
}
